// SoftEmbRescore_9577777070758
// MI455X (gfx1250) — matrix operand order repaired and checked by reading, not compiled, not run on hardware
//
#include <hip/hip_runtime.h>
#include <math.h>

typedef __attribute__((ext_vector_type(16))) _Float16 v16h;
typedef __attribute__((ext_vector_type(8)))  float    v8f;

#define NPAIR 65536
#define NP    1024          // N_PRED == EMB_DIM
#define EPSV  1e-8f

// split f32 into f16 hi + f16 lo (residual): a ~= hi + lo with ~2^-22 rel. error
__device__ __forceinline__ void splitf(float f, _Float16& h, _Float16& l) {
    _Float16 hh = (_Float16)f;
    h = hh;
    l = (_Float16)(f - (float)hh);
}

// A fragment: 16x32 (MxK) f16 split on the fly from row-major f32 src
// (already offset to row0,k0).  A layout: lane L -> row L&15; lanes 0-15
// hold K 0..15, lanes 16-31 hold K 16..31 (matches B fragment order).
__device__ __forceinline__ void load_frag_a(const float* __restrict__ src, int ld,
                                            v16h& hi, v16h& lo) {
    const int lane = threadIdx.x & 31;
    const int r    = lane & 15;
    const int kh   = lane >> 4;                 // 0 or 1
    const float4* p0 = (const float4*)(src + r * ld + kh * 16);
    float4 f0 = p0[0], f1 = p0[1], f2 = p0[2], f3 = p0[3];
    float v[16] = {f0.x,f0.y,f0.z,f0.w, f1.x,f1.y,f1.z,f1.w,
                   f2.x,f2.y,f2.z,f2.w, f3.x,f3.y,f3.z,f3.w};
#pragma unroll
    for (int i = 0; i < 16; i++) { _Float16 h,l; splitf(v[i],h,l); hi[i]=h; lo[i]=l; }
}

// Same as load_frag_a but for the gram kernel's B operand (columns of E^T ==
// rows of E) in the same K order as A: lane -> column L&15; lanes 0-15 hold K 0..15,
// lanes 16-31 hold K 16..31 (16 consecutive K per half-wave).
__device__ __forceinline__ void load_frag_bT_f32(const float* __restrict__ src, int ld,
                                                 v16h& hi, v16h& lo) {
    const int lane = threadIdx.x & 31;
    const int n    = lane & 15;
    const int kh   = lane >> 4;
    const float4* p = (const float4*)(src + n * ld + kh * 16);
    float4 f0 = p[0], f1 = p[1], f2 = p[2], f3 = p[3];
    float v[16] = {f0.x,f0.y,f0.z,f0.w, f1.x,f1.y,f1.z,f1.w,
                   f2.x,f2.y,f2.z,f2.w, f3.x,f3.y,f3.z,f3.w};
#pragma unroll
    for (int i = 0; i < 16; i++) { _Float16 h,l; splitf(v[i],h,l); hi[i]=h; lo[i]=l; }
}

// B fragment straight from a PRE-SPLIT row-major f16 matrix whose rows are the
// B columns (G is symmetric).  16 consecutive halves = 32 bytes per lane ->
// two global_load_b128, zero conversion VALU in the hot loop.
__device__ __forceinline__ v16h load_frag_b_h(const _Float16* __restrict__ src, int ld) {
    const int lane = threadIdx.x & 31;
    const int n    = lane & 15;
    const int kh   = lane >> 4;
    return *(const v16h*)(src + n * ld + kh * 16);
}

// near-fp32 product via 3 WMMAs: ah*bh + ah*bl + al*bh (drop al*bl ~2^-22)
__device__ __forceinline__ v8f wmma3(v8f c, v16h ah, v16h al, v16h bh, v16h bl) {
    c = __builtin_amdgcn_wmma_f32_16x16x32_f16(false, ah, false, bh, (short)0, c, false, false);
    c = __builtin_amdgcn_wmma_f32_16x16x32_f16(false, ah, false, bl, (short)0, c, false, false);
    c = __builtin_amdgcn_wmma_f32_16x16x32_f16(false, al, false, bh, (short)0, c, false, false);
    return c;
}

// ---------------------------------------------------------------------------
// Kernel 1: G = E * E^T, stored directly in SPLIT f16 form (Gh + Gl).
// One wave per 16x16 tile, 4 waves/block.  B columns of E^T == rows of E.
__global__ __launch_bounds__(128) void gram_kernel(const float* __restrict__ emb,
                                                   _Float16* __restrict__ Gh,
                                                   _Float16* __restrict__ Gl) {
    const int wave = threadIdx.x >> 5;
    const int tile = blockIdx.x * 4 + wave;           // 0..4095
    const int mt   = (tile >> 6) << 4;
    const int nt   = (tile & 63) << 4;

    v8f acc = {};
    for (int k = 0; k < NP; k += 32) {
        v16h ah, al, bh, bl;
        load_frag_a     (emb + (size_t)mt * NP + k, NP, ah, al);
        load_frag_bT_f32(emb + (size_t)nt * NP + k, NP, bh, bl);
        acc = wmma3(acc, ah, al, bh, bl);
    }
    const int lane  = threadIdx.x & 31;
    const int col   = nt + (lane & 15);
    const int rbase = mt + ((lane >> 4) << 3);
#pragma unroll
    for (int v = 0; v < 8; v++) {
        _Float16 h, l;
        splitf(acc[v], h, l);
        Gh[(size_t)(rbase + v) * NP + col] = h;
        Gl[(size_t)(rbase + v) * NP + col] = l;
    }
}

// Kernel 1b: embn_j = ||E_j||2 (1024 values; trivial traffic)
__global__ __launch_bounds__(256) void norm_kernel(const float* __restrict__ emb,
                                                   float* __restrict__ embn) {
    const int j = blockIdx.x * 256 + threadIdx.x;
    const float* row = emb + (size_t)j * NP;
    float s = 0.f;
    for (int k = 0; k < NP; k++) { float f = row[k]; s += f * f; }
    embn[j] = sqrtf(s);
}

// ---------------------------------------------------------------------------
// Kernel 2: D = X * G (G pre-split f16), fused with soft_norm_i =
// sqrt(dot(D_i, x_i)) and the cosine scaling.  Block = 512 threads (16 waves)
// computes 32 full rows: wave w -> rows base+(w>>3)*16.., cols (w&7)*128..
__global__ __launch_bounds__(512) void cos_kernel(const float* __restrict__ x,
                                                  const _Float16* __restrict__ Gh,
                                                  const _Float16* __restrict__ Gl,
                                                  const float* __restrict__ embn,
                                                  float* __restrict__ out) {
    __shared__ float lds_rd[32][8];
    __shared__ float lds_n[32];

    const int wave  = threadIdx.x >> 5;
    const int lane  = threadIdx.x & 31;
    const int mhalf = wave >> 3;                       // 0/1
    const int nw    = wave & 7;                        // N-span index
    const int row0  = (blockIdx.x << 5) + (mhalf << 4);
    const int col0  = nw << 7;

    v8f acc[8] = {};
    for (int k = 0; k < NP; k += 32) {
        v16h ah, al;
        load_frag_a(x + (size_t)row0 * NP + k, NP, ah, al);
#pragma unroll
        for (int t = 0; t < 8; t++) {
            // G symmetric: column c of G == row c of G -> contiguous f16 loads
            const size_t off = (size_t)(col0 + t * 16) * NP + k;
            v16h bh = load_frag_b_h(Gh + off, NP);
            v16h bl = load_frag_b_h(Gl + off, NP);
            acc[t]  = wmma3(acc[t], ah, al, bh, bl);
        }
    }

    // C layout: lane -> col = lane&15; VGPR v -> row = v + 8*(lane>>4)
    const int kh = lane >> 4;
    const int cl = lane & 15;

    // partial dot(D_i, x_i) over this wave's 128 columns
    float pd[8];
#pragma unroll
    for (int v = 0; v < 8; v++) pd[v] = 0.f;
#pragma unroll
    for (int t = 0; t < 8; t++) {
        const int c = col0 + t * 16 + cl;
#pragma unroll
        for (int v = 0; v < 8; v++) {
            const int r = row0 + (kh << 3) + v;
            pd[v] += acc[t][v] * x[(size_t)r * NP + c];
        }
    }
    // reduce across the 16 lanes of each half-wave (xor masks stay in-half)
#pragma unroll
    for (int m = 8; m >= 1; m >>= 1)
#pragma unroll
        for (int v = 0; v < 8; v++) pd[v] += __shfl_xor(pd[v], m, 32);

    if ((lane & 15) == 0) {
        const int rl = (mhalf << 4) + (kh << 3);
#pragma unroll
        for (int v = 0; v < 8; v++) lds_rd[rl + v][nw] = pd[v];
    }
    __syncthreads();
    if (threadIdx.x < 32) {
        float s = 0.f;
#pragma unroll
        for (int t = 0; t < 8; t++) s += lds_rd[threadIdx.x][t];
        lds_n[threadIdx.x] = sqrtf(s);
    }
    __syncthreads();

    // cos = D / max(soft_norm * emb_norm, eps); store 32x1024 block
#pragma unroll
    for (int t = 0; t < 8; t++) {
        const int   c  = col0 + t * 16 + cl;
        const float en = embn[c];
#pragma unroll
        for (int v = 0; v < 8; v++) {
            const int   rl    = (mhalf << 4) + (kh << 3) + v;
            const float denom = fmaxf(lds_n[rl] * en, EPSV);
            out[(size_t)(row0 + (kh << 3) + v) * NP + c] = acc[t][v] / denom;
        }
    }
}

// ---------------------------------------------------------------------------
extern "C" void kernel_launch(void* const* d_in, const int* in_sizes, int n_in,
                              void* d_out, int out_size, void* d_ws, size_t ws_size,
                              hipStream_t stream) {
    const float* x   = (const float*)d_in[0];   // [1, 65536, 1024] f32
    const float* emb = (const float*)d_in[1];   // [1024, 1024]     f32
    float*       out = (float*)d_out;           // [1, 65536, 1024] f32

    // workspace: Gh (2MB) | Gl (2MB) | embn (4KB)   -> ~4 MB total
    _Float16* Gh   = (_Float16*)d_ws;
    _Float16* Gl   = (_Float16*)((char*)d_ws + (size_t)NP * NP * sizeof(_Float16));
    float*    embn = (float*)  ((char*)d_ws + (size_t)2 * NP * NP * sizeof(_Float16));

    gram_kernel<<<(NP / 16) * (NP / 16) / 4, 128, 0, stream>>>(emb, Gh, Gl);
    norm_kernel<<<NP / 256, 256, 0, stream>>>(emb, embn);
    cos_kernel<<<NPAIR / 32, 512, 0, stream>>>(x, Gh, Gl, embn, out);
}
